// CSJ_RQVAE_31602369364527
// MI455X (gfx1250) — matrix operand order repaired and checked by reading, compiled, not run on hardware
//
#include <hip/hip_runtime.h>
#include <hip/hip_bf16.h>
#include <math.h>

// ---------------------------------------------------------------------------
// RQ-VAE forward, MI455X (gfx1250, wave32, WMMA).
// GEMMs use v_wmma_f32_16x16x32_bf16 (bf16 in, fp32 accum). LN stats, GELU,
// residual updates, codebook gathers and losses stay fp32.
// Round-2 changes: LDS-staged encoder inputs (ds_load fragments, global_load
// weights), no wide fragment hoisting in the encoder, #pragma unroll 1 on
// N-tile/chunk/level loops to prevent LICM register bloat + scratch spills,
// direct pointer provenance (no pointer-array selects) for global loads.
// ---------------------------------------------------------------------------

#define BATCH 65536
#define DIN   384
#define HDIM  256
#define NLVL  4
#define NCODE 1024
#define MT    32          // rows per block
#define NTHR  256         // 8 waves

typedef __attribute__((ext_vector_type(16))) __bf16 v16bf;
typedef __attribute__((ext_vector_type(8)))  __bf16 v8bf;
typedef __attribute__((ext_vector_type(8)))  float  v8f;

__device__ __forceinline__ v8f wmma_bf16(v16bf a, v16bf b, v8f c) {
  return __builtin_amdgcn_wmma_f32_16x16x32_bf16(false, a, false, b, (short)0, c,
                                                 false, false);
}

// A fragment (16x32 bf16, MxK): lane m = lane&15; lane holds K chunks
// [k0..k0+7] (VGPR0-3) and [k0+16..k0+23] (VGPR4-7) with k0 = 8*(lane>>4).
__device__ __forceinline__ v16bf load_a_frag_bf(const __bf16* base, int lda) {
  int lane = threadIdx.x & 31;
  const __bf16* p = base + (lane & 15) * lda + ((lane >> 4) << 3);
  v16bf a;
  *(v8bf*)&a       = *(const v8bf*)p;
  *((v8bf*)&a + 1) = *(const v8bf*)(p + 16);
  return a;
}

// B fragment (32x16 bf16, KxN) from N-major storage (row n = column of B,
// ldb = K stride): lane n = lane&15 holds K chunks [k0..k0+7] and
// [k0+16..k0+23] with k0 = 8*(lane>>4).
__device__ __forceinline__ v16bf load_b_frag(const __bf16* base, int ldb) {
  int lane = threadIdx.x & 31;
  const __bf16* p = base + (lane & 15) * ldb + ((lane >> 4) << 3);
  v16bf b;
  *(v8bf*)&b       = *(const v8bf*)p;
  *((v8bf*)&b + 1) = *(const v8bf*)(p + 16);
  return b;
}

// LayerNorm(+optional exact-erf GELU) over MT x 256 fp32 LDS tile -> bf16 tile.
__device__ __forceinline__ void ln_gelu_bf(const float (*act)[HDIM],
                                           __bf16 (*obf)[HDIM],
                                           const float* g, const float* be,
                                           float (*pm)[8], float (*pv)[8],
                                           bool do_gelu) {
  int row = threadIdx.x >> 3;   // 0..31
  int seg = threadIdx.x & 7;    // 8 threads/row, 32 elems each
  float s = 0.f, ss = 0.f;
#pragma unroll
  for (int i = 0; i < 32; ++i) { float v = act[row][seg * 32 + i]; s += v; ss += v * v; }
  pm[row][seg] = s; pv[row][seg] = ss;
  __syncthreads();
  float mean = 0.f, msq = 0.f;
#pragma unroll
  for (int i = 0; i < 8; ++i) { mean += pm[row][i]; msq += pv[row][i]; }
  mean *= (1.f / HDIM);
  float var = msq * (1.f / HDIM) - mean * mean;
  float rstd = rsqrtf(var + 1e-5f);
#pragma unroll
  for (int i = 0; i < 32; ++i) {
    int c = seg * 32 + i;
    float v = (act[row][c] - mean) * rstd * g[c] + be[c];
    if (do_gelu) v = 0.5f * v * (1.f + erff(v * 0.70710678118654752f));
    obf[row][c] = (__bf16)v;
  }
}

// ---------------------------------------------------------------------------
// Prep kernels
// ---------------------------------------------------------------------------
__global__ void k_transpose_bf(const float* __restrict__ W, __bf16* __restrict__ WT,
                               int K, int N) {
  int i = blockIdx.x * NTHR + threadIdx.x;
  if (i < K * N) {
    int k = i / N, n = i % N;
    WT[(size_t)n * K + k] = (__bf16)W[i];
  }
}

__global__ __launch_bounds__(NTHR) void k_cb_prep(const float* __restrict__ cb,
                                                  __bf16* __restrict__ cbbf,
                                                  float* __restrict__ cnorm) {
  int code = blockIdx.x;                       // 0..4095 (level*1024 + idx)
  __shared__ float red[NTHR];
  float v = cb[(size_t)code * HDIM + threadIdx.x];
  cbbf[(size_t)code * HDIM + threadIdx.x] = (__bf16)v;
  red[threadIdx.x] = v * v;
  __syncthreads();
  for (int s = 128; s; s >>= 1) {
    if (threadIdx.x < s) red[threadIdx.x] += red[threadIdx.x + s];
    __syncthreads();
  }
  if (threadIdx.x == 0) cnorm[code] = red[0];
}

__global__ void k_init(float* acc) { if (threadIdx.x < 3) acc[threadIdx.x] = 0.f; }

// ---------------------------------------------------------------------------
// Fused encoder: h_joint = 0.5*enc(z_sem) + 0.5*enc(z_cf)
// ---------------------------------------------------------------------------
__global__ __launch_bounds__(NTHR) void k_encode(
    const float* __restrict__ z_sem, const float* __restrict__ z_cf,
    const __bf16* __restrict__ w1T, const float* __restrict__ b1,
    const float* __restrict__ g1, const float* __restrict__ be1,
    const __bf16* __restrict__ w2T, const float* __restrict__ b2,
    const float* __restrict__ g2, const float* __restrict__ be2,
    float* __restrict__ h_out) {
  __shared__ __bf16 s_in[MT][DIN];    // staged input tile (bf16)
  __shared__ float  s_act[MT][HDIM];
  __shared__ __bf16 s_bf[MT][HDIM];
  __shared__ float  s_h[MT][HDIM];
  __shared__ float  s_m[MT][8], s_v[MT][8];

  const int rowBase = blockIdx.x * MT;
  const int wave = threadIdx.x >> 5, lane = threadIdx.x & 31;
  const int rt = wave & 1;          // row tile (16 rows)
  const int cg = wave >> 1;         // col group
  const int mb = rt * 16 + ((lane >> 4) << 3);
  const int nlo = lane & 15;

  auto do_pass = [&](const float* __restrict__ x, bool first) {
    // ---- stage input tile to bf16 LDS (direct global provenance) ----
    {
      int row = threadIdx.x >> 3, seg = threadIdx.x & 7;   // 8 thr/row, 48 ea
      const float* src = x + (size_t)(rowBase + row) * DIN + seg * 48;
#pragma unroll
      for (int i = 0; i < 48; ++i) s_in[row][seg * 48 + i] = (__bf16)src[i];
    }
    __syncthreads();
    // ---- layer 1: [MT x 384] @ [384 x 256] ----
#pragma unroll 1
    for (int j = 0; j < 4; ++j) {
      int nt = cg * 4 + j;                     // 0..15
      v8f acc = {};
#pragma unroll
      for (int kk = 0; kk < DIN / 32; ++kk) {
        v16bf a = load_a_frag_bf(&s_in[rt * 16][0] + kk * 32, DIN);
        v16bf b = load_b_frag(w1T + (size_t)nt * 16 * DIN + kk * 32, DIN);
        acc = wmma_bf16(a, b, acc);
      }
      int n = nt * 16 + nlo;
      float bias = b1[n];
#pragma unroll
      for (int r = 0; r < 8; ++r) s_act[mb + r][n] = acc[r] + bias;
    }
    __syncthreads();
    ln_gelu_bf(s_act, s_bf, g1, be1, s_m, s_v, true);
    __syncthreads();
    // ---- layer 2: [MT x 256] @ [256 x 256] ----
#pragma unroll 1
    for (int j = 0; j < 4; ++j) {
      int nt = cg * 4 + j;
      v8f acc = {};
#pragma unroll
      for (int kk = 0; kk < HDIM / 32; ++kk) {
        v16bf a = load_a_frag_bf(&s_bf[rt * 16][0] + kk * 32, HDIM);
        v16bf b = load_b_frag(w2T + (size_t)nt * 16 * HDIM + kk * 32, HDIM);
        acc = wmma_bf16(a, b, acc);
      }
      int n = nt * 16 + nlo;
      float bias = b2[n];
#pragma unroll
      for (int r = 0; r < 8; ++r) s_act[mb + r][n] = acc[r] + bias;
    }
    __syncthreads();
    // ---- LN2 (no gelu), accumulate 0.5*val into s_h ----
    {
      int row = threadIdx.x >> 3, seg = threadIdx.x & 7;
      float s = 0.f, ss = 0.f;
#pragma unroll
      for (int i = 0; i < 32; ++i) { float v = s_act[row][seg * 32 + i]; s += v; ss += v * v; }
      s_m[row][seg] = s; s_v[row][seg] = ss;
      __syncthreads();
      float mean = 0.f, msq = 0.f;
#pragma unroll
      for (int i = 0; i < 8; ++i) { mean += s_m[row][i]; msq += s_v[row][i]; }
      mean *= (1.f / HDIM);
      float rstd = rsqrtf(msq * (1.f / HDIM) - mean * mean + 1e-5f);
#pragma unroll
      for (int i = 0; i < 32; ++i) {
        int c = seg * 32 + i;
        float v = (s_act[row][c] - mean) * rstd * g2[c] + be2[c];
        float contrib = 0.5f * v;                // LAM = 0.5 for both
        if (first) s_h[row][c] = contrib; else s_h[row][c] += contrib;
      }
    }
    __syncthreads();
  };
  do_pass(z_sem, true);
  do_pass(z_cf, false);

  for (int t = threadIdx.x; t < MT * HDIM; t += NTHR) {
    int r = t >> 8, c = t & 255;
    h_out[(size_t)(rowBase + r) * HDIM + c] = s_h[r][c];
  }
}

// ---------------------------------------------------------------------------
// Residual quantizer: 4 levels of WMMA scoring + argmin + fp32 residual update
// ---------------------------------------------------------------------------
__global__ __launch_bounds__(NTHR) void k_rq(
    const float* __restrict__ h, const __bf16* __restrict__ cb_bf,
    const float* __restrict__ cb_f32, const float* __restrict__ cnorm,
    __bf16* __restrict__ quant_bf, float* __restrict__ codes_out,
    float* __restrict__ commit_accum) {
  __shared__ float  s_rf[MT][HDIM];
  __shared__ __bf16 s_rb[MT][HDIM];
  __shared__ float  s_bv[4][MT];
  __shared__ int    s_bi[4][MT];
  __shared__ int    s_idx[MT];
  __shared__ float  s_red[NTHR];

  const int rowBase = blockIdx.x * MT;
  const int wave = threadIdx.x >> 5, lane = threadIdx.x & 31;
  const int rt = wave & 1;         // row tile
  const int cg = wave >> 1;        // 256 codes per col group
  const int mb = rt * 16 + ((lane >> 4) << 3);
  const int nlo = lane & 15;

  for (int t = threadIdx.x; t < MT * HDIM; t += NTHR) {
    int r = t >> 8, c = t & 255;
    float v = h[(size_t)(rowBase + r) * HDIM + c];
    s_rf[r][c] = v; s_rb[r][c] = (__bf16)v;
  }
  float commit = 0.f;
  __syncthreads();

#pragma unroll 1
  for (int l = 0; l < NLVL; ++l) {
    const __bf16* cbl = cb_bf + (size_t)l * NCODE * HDIM;
    const float*  cnl = cnorm + l * NCODE;
    // Hoist A fragments for this level: 8 * v16bf = 64 VGPRs, reused 16x.
    v16bf afr[HDIM / 32];
#pragma unroll
    for (int kk = 0; kk < HDIM / 32; ++kk)
      afr[kk] = load_a_frag_bf(&s_rb[rt * 16][0] + kk * 32, HDIM);

    float bv[8]; int bi[8];
#pragma unroll
    for (int r = 0; r < 8; ++r) { bv[r] = 3.4e38f; bi[r] = 0x7fffffff; }

#pragma unroll 1
    for (int ch = 0; ch < 16; ++ch) {            // 16 chunks of 16 codes
      int n0 = cg * 256 + ch * 16;
      v8f acc = {};
#pragma unroll
      for (int kk = 0; kk < HDIM / 32; ++kk)
        acc = wmma_bf16(afr[kk], load_b_frag(cbl + (size_t)n0 * HDIM + kk * 32, HDIM), acc);
      int n = n0 + nlo;
      float cn = cnl[n];
#pragma unroll
      for (int r = 0; r < 8; ++r) {
        float sc = cn - 2.f * acc[r];            // == ||c||^2 - 2 r.c
        if (sc < bv[r] || (sc == bv[r] && n < bi[r])) { bv[r] = sc; bi[r] = n; }
      }
    }
    // reduce across the 16 lanes holding the same row (N direction)
#pragma unroll
    for (int r = 0; r < 8; ++r) {
#pragma unroll
      for (int mask = 8; mask; mask >>= 1) {
        float ov = __shfl_xor(bv[r], mask, 16);
        int   oi = __shfl_xor(bi[r], mask, 16);
        if (ov < bv[r] || (ov == bv[r] && oi < bi[r])) { bv[r] = ov; bi[r] = oi; }
      }
    }
    if (nlo == 0) {
#pragma unroll
      for (int r = 0; r < 8; ++r) { s_bv[cg][mb + r] = bv[r]; s_bi[cg][mb + r] = bi[r]; }
    }
    __syncthreads();
    if (threadIdx.x < MT) {
      int row = threadIdx.x;
      float best = s_bv[0][row]; int bidx = s_bi[0][row];
#pragma unroll
      for (int c = 1; c < 4; ++c) {
        float v = s_bv[c][row]; int i2 = s_bi[c][row];
        if (v < best || (v == best && i2 < bidx)) { best = v; bidx = i2; }
      }
      s_idx[row] = bidx;
      codes_out[(size_t)(rowBase + row) * NLVL + l] = (float)bidx;
    }
    __syncthreads();
    // fp32 residual update with fp32 codebook gather
    {
      int row = threadIdx.x >> 3, seg = threadIdx.x & 7;
      const float* q = cb_f32 + ((size_t)l * NCODE + s_idx[row]) * HDIM + seg * 32;
#pragma unroll
      for (int i = 0; i < 32; ++i) {
        int c = seg * 32 + i;
        float rn = s_rf[row][c] - q[i];
        s_rf[row][c] = rn;
        s_rb[row][c] = (__bf16)rn;
        commit += rn * rn;                       // mean((r - q)^2) == mean(r_new^2)
      }
    }
    __syncthreads();
  }
  // quantized = h - final residual
  for (int t = threadIdx.x; t < MT * HDIM; t += NTHR) {
    int r = t >> 8, c = t & 255;
    float v = h[(size_t)(rowBase + r) * HDIM + c] - s_rf[r][c];
    quant_bf[(size_t)(rowBase + r) * HDIM + c] = (__bf16)v;
  }
  s_red[threadIdx.x] = commit;
  __syncthreads();
  for (int s = 128; s; s >>= 1) {
    if (threadIdx.x < s) s_red[threadIdx.x] += s_red[threadIdx.x + s];
    __syncthreads();
  }
  if (threadIdx.x == 0) atomicAdd(commit_accum, s_red[0]);
}

// ---------------------------------------------------------------------------
// Fused decoder + MSE accumulation (launched once per decoder)
// ---------------------------------------------------------------------------
__global__ __launch_bounds__(NTHR) void k_decode(
    const __bf16* __restrict__ qbf, const __bf16* __restrict__ w1T,
    const float* __restrict__ b1, const float* __restrict__ g1,
    const float* __restrict__ be1, const __bf16* __restrict__ w2T,
    const float* __restrict__ b2, const float* __restrict__ z,
    float* __restrict__ recon, float* __restrict__ sse_accum) {
  __shared__ float  s_act[MT][HDIM];
  __shared__ __bf16 s_bf[MT][HDIM];
  __shared__ float  s_m[MT][8], s_v[MT][8];
  __shared__ float  s_red[NTHR];

  const int rowBase = blockIdx.x * MT;
  const int wave = threadIdx.x >> 5, lane = threadIdx.x & 31;
  const int rt = wave & 1, cg = wave >> 1;
  const int mb = rt * 16 + ((lane >> 4) << 3);
  const int nlo = lane & 15;

  // ---- layer 1: [MT x 256] @ [256 x 256], A frags from global bf16 ----
  {
    v16bf afr[HDIM / 32];      // 64 VGPRs, reused across 4 N tiles
#pragma unroll
    for (int kk = 0; kk < HDIM / 32; ++kk)
      afr[kk] = load_a_frag_bf(qbf + (size_t)(rowBase + rt * 16) * HDIM + kk * 32, HDIM);
#pragma unroll 1
    for (int j = 0; j < 4; ++j) {
      int nt = cg * 4 + j;
      v8f acc = {};
#pragma unroll
      for (int kk = 0; kk < HDIM / 32; ++kk)
        acc = wmma_bf16(afr[kk], load_b_frag(w1T + (size_t)nt * 16 * HDIM + kk * 32, HDIM), acc);
      int n = nt * 16 + nlo;
      float bias = b1[n];
#pragma unroll
      for (int r = 0; r < 8; ++r) s_act[mb + r][n] = acc[r] + bias;
    }
  }
  __syncthreads();
  ln_gelu_bf(s_act, s_bf, g1, be1, s_m, s_v, true);
  __syncthreads();
  // ---- layer 2: [MT x 256] @ [256 x 384] + MSE vs z, recon to d_out ----
  float sse = 0.f;
  {
    v16bf afr[HDIM / 32];
#pragma unroll
    for (int kk = 0; kk < HDIM / 32; ++kk)
      afr[kk] = load_a_frag_bf(&s_bf[rt * 16][0] + kk * 32, HDIM);
#pragma unroll 1
    for (int j = 0; j < 6; ++j) {
      int nt = cg * 6 + j;                       // 0..23 (N = 384)
      v8f acc = {};
#pragma unroll
      for (int kk = 0; kk < HDIM / 32; ++kk)
        acc = wmma_bf16(afr[kk], load_b_frag(w2T + (size_t)nt * 16 * HDIM + kk * 32, HDIM), acc);
      int n = nt * 16 + nlo;
      float bias = b2[n];
#pragma unroll
      for (int r = 0; r < 8; ++r) {
        size_t gi = (size_t)(rowBase + mb + r) * DIN + n;
        float v = acc[r] + bias;
        float d = v - z[gi];
        sse += d * d;
        recon[gi] = v;
      }
    }
  }
  s_red[threadIdx.x] = sse;
  __syncthreads();
  for (int s = 128; s; s >>= 1) {
    if (threadIdx.x < s) s_red[threadIdx.x] += s_red[threadIdx.x + s];
    __syncthreads();
  }
  if (threadIdx.x == 0) atomicAdd(sse_accum, s_red[0]);
}

__global__ void k_final(const float* __restrict__ acc, float* __restrict__ out0) {
  // acc[0]=commit SSE, acc[1]=sem SSE, acc[2]=cf SSE
  float commit = acc[0] / ((float)NLVL * (float)BATCH * (float)HDIM);
  float recon = 0.5f * acc[1] / ((float)BATCH * (float)DIN)
              + 0.5f * acc[2] / ((float)BATCH * (float)DIN);
  *out0 = recon + commit;
}

// ---------------------------------------------------------------------------
extern "C" void kernel_launch(void* const* d_in, const int* in_sizes, int n_in,
                              void* d_out, int out_size, void* d_ws, size_t ws_size,
                              hipStream_t stream) {
  // Inputs (setup_inputs dict order, nested dicts flattened in insertion order)
  const float* z_sem = (const float*)d_in[0];
  const float* z_cf  = (const float*)d_in[1];
  const float* e_w1  = (const float*)d_in[2];
  const float* e_b1  = (const float*)d_in[3];
  const float* e_g1  = (const float*)d_in[4];
  const float* e_be1 = (const float*)d_in[5];
  const float* e_w2  = (const float*)d_in[6];
  const float* e_b2  = (const float*)d_in[7];
  const float* e_g2  = (const float*)d_in[8];
  const float* e_be2 = (const float*)d_in[9];
  const float* ds_w1 = (const float*)d_in[10];
  const float* ds_b1 = (const float*)d_in[11];
  const float* ds_g1 = (const float*)d_in[12];
  const float* ds_be1= (const float*)d_in[13];
  const float* ds_w2 = (const float*)d_in[14];
  const float* ds_b2 = (const float*)d_in[15];
  const float* dc_w1 = (const float*)d_in[16];
  const float* dc_b1 = (const float*)d_in[17];
  const float* dc_g1 = (const float*)d_in[18];
  const float* dc_be1= (const float*)d_in[19];
  const float* dc_w2 = (const float*)d_in[20];
  const float* dc_b2 = (const float*)d_in[21];
  const float* cb    = (const float*)d_in[22];

  // Workspace carve-up (~104 MB total)
  char* ws = (char*)d_ws;
  size_t off = 0;
  auto take = [&](size_t bytes) { char* p = ws + off; off = (off + bytes + 255) & ~(size_t)255; return p; };
  __bf16* cb_bf   = (__bf16*)take((size_t)NLVL * NCODE * HDIM * 2);
  float*  cnorm   = (float*) take((size_t)NLVL * NCODE * 4);
  __bf16* ew1T    = (__bf16*)take((size_t)HDIM * DIN * 2);
  __bf16* ew2T    = (__bf16*)take((size_t)HDIM * HDIM * 2);
  __bf16* dsw1T   = (__bf16*)take((size_t)HDIM * HDIM * 2);
  __bf16* dsw2T   = (__bf16*)take((size_t)DIN * HDIM * 2);
  __bf16* dcw1T   = (__bf16*)take((size_t)HDIM * HDIM * 2);
  __bf16* dcw2T   = (__bf16*)take((size_t)DIN * HDIM * 2);
  float*  h_joint = (float*) take((size_t)BATCH * HDIM * 4);
  __bf16* qbf     = (__bf16*)take((size_t)BATCH * HDIM * 2);
  float*  accums  = (float*) take(256); // [0]=commit, [1]=sse_sem, [2]=sse_cf

  // d_out layout: [loss(1)][sem_recon(B*DIN)][cf_recon(B*DIN)][codes(B*4)]
  float* out       = (float*)d_out;
  float* out_sem   = out + 1;
  float* out_cf    = out + 1 + (size_t)BATCH * DIN;
  float* out_codes = out + 1 + (size_t)2 * BATCH * DIN;

  // --- prep ---
  hipLaunchKernelGGL(k_init, dim3(1), dim3(64), 0, stream, accums);
  hipLaunchKernelGGL(k_cb_prep, dim3(NLVL * NCODE), dim3(NTHR), 0, stream, cb, cb_bf, cnorm);
  auto tr = [&](const float* W, __bf16* WT, int K, int N) {
    hipLaunchKernelGGL(k_transpose_bf, dim3((K * N + NTHR - 1) / NTHR), dim3(NTHR), 0, stream, W, WT, K, N);
  };
  tr(e_w1, ew1T, DIN, HDIM);
  tr(e_w2, ew2T, HDIM, HDIM);
  tr(ds_w1, dsw1T, HDIM, HDIM);
  tr(ds_w2, dsw2T, HDIM, DIN);
  tr(dc_w1, dcw1T, HDIM, HDIM);
  tr(dc_w2, dcw2T, HDIM, DIN);

  const int nblk = BATCH / MT;   // 2048
  // --- encoder (both inputs fused, shared weights) ---
  hipLaunchKernelGGL(k_encode, dim3(nblk), dim3(NTHR), 0, stream,
                     z_sem, z_cf, ew1T, e_b1, e_g1, e_be1, ew2T, e_b2, e_g2, e_be2,
                     h_joint);
  // --- residual quantization ---
  hipLaunchKernelGGL(k_rq, dim3(nblk), dim3(NTHR), 0, stream,
                     h_joint, cb_bf, cb, cnorm, qbf, out_codes, &accums[0]);
  // --- decoders ---
  hipLaunchKernelGGL(k_decode, dim3(nblk), dim3(NTHR), 0, stream,
                     qbf, dsw1T, ds_b1, ds_g1, ds_be1, dsw2T, ds_b2,
                     z_sem, out_sem, &accums[1]);
  hipLaunchKernelGGL(k_decode, dim3(nblk), dim3(NTHR), 0, stream,
                     qbf, dcw1T, dc_b1, dc_g1, dc_be1, dcw2T, dc_b2,
                     z_cf, out_cf, &accums[2]);
  // --- scalar loss ---
  hipLaunchKernelGGL(k_final, dim3(1), dim3(1), 0, stream, accums, out);
}
